// CertViT_17446157156739
// MI455X (gfx1250) — matrix operand order repaired and checked by reading, not compiled, not run on hardware
//
#include <hip/hip_runtime.h>
#include <math.h>

// ---------------------------------------------------------------------------
// CDNA5 (gfx1250) ViT forward: bf16 WMMA GEMMs, f32 accumulation,
// bf16-resident weights/activations, double-buffered async Global->LDS staging,
// bf16 attention probabilities + transposed-V cache for fully vector WMMA feeds.
// ---------------------------------------------------------------------------

#define USE_ASYNC_STAGING 1

typedef __attribute__((ext_vector_type(16))) __bf16 v16bf;
typedef __attribute__((ext_vector_type(8)))  float  v8f;

#define WMMA_BF16(A_, B_, C_) \
  __builtin_amdgcn_wmma_f32_16x16x32_bf16(false, (A_), false, (B_), (short)0, (C_), false, false)

static constexpr int BATCH  = 64;
static constexpr int Cdim   = 768;
static constexpr int C3     = 2304;   // 3*C
static constexpr int C4     = 3072;   // 4*C
static constexpr int NH     = 12;     // heads
static constexpr int DH     = 64;     // head dim
static constexpr int NP     = 224;    // padded token count (multiple of 32)
static constexpr int NPATCH = 196;
static constexpr int N0     = 197;    // tokens before pruning
static constexpr int N1     = 139;    // tokens after pruning (1 + 137 + 1)
static constexpr int KSEL   = 137;
static constexpr int NREJ   = 59;     // 196 - 137
static constexpr int Mrows  = BATCH * NP;  // 14336, multiple of 64

// ---- bf16 convert (round-to-nearest-even) ---------------------------------
__device__ __forceinline__ __bf16 f2bf(float f) {
  unsigned int u = __float_as_uint(f);
  unsigned int r = (u + 0x7FFFu + ((u >> 16) & 1u)) >> 16;
  unsigned short s = (unsigned short)r;
  __bf16 b;
  __builtin_memcpy(&b, &s, 2);
  return b;
}

// ---- async 16B global -> LDS copy (CDNA5, tracked by ASYNCcnt) ------------
__device__ __forceinline__ void async_copy16(const void* gptr, void* lptr) {
#if USE_ASYNC_STAGING
  unsigned int loff = (unsigned int)(unsigned long long)lptr;  // LDS byte offset
  asm volatile("global_load_async_to_lds_b128 %0, %1, off"
               :: "v"(loff), "v"(gptr) : "memory");
#else
  *(uint4*)lptr = *(const uint4*)gptr;
#endif
}
template<int N>
__device__ __forceinline__ void async_wait() {
#if USE_ASYNC_STAGING
  asm volatile("s_wait_asynccnt %0" :: "i"(N) : "memory");
#endif
}

// ---- fragment loaders ------------------------------------------------------
// A (16x32 bf16): lane half 0 rows hold K = k0+[0..7] and k0+[16..23];
//                 lane half 1 rows hold K = k0+[8..15] and k0+[24..31].
// B (32x16 bf16): lane holds one column; same K split as A (k0+half*8+[0..7], k0+16+half*8+[0..7]).
__device__ __forceinline__ v16bf load_a_frag_bf(const __bf16* __restrict__ row, int k0, int half) {
  union { uint4 u[2]; v16bf v; } t;
  t.u[0] = *(const uint4*)(row + k0 + half * 8);
  t.u[1] = *(const uint4*)(row + k0 + 16 + half * 8);
  return t.v;
}
__device__ __forceinline__ v16bf load_b_frag_bf(const __bf16* __restrict__ row, int k0, int half) {
  union { uint4 u[2]; v16bf v; } t;
  t.u[0] = *(const uint4*)(row + k0 + half * 8);
  t.u[1] = *(const uint4*)(row + k0 + 16 + half * 8);
  return t.v;
}
__device__ __forceinline__ v16bf load_a_frag_lds(const __bf16* base, int half) {
  union { uint4 u[2]; v16bf v; } t;
  t.u[0] = *(const uint4*)(base + half * 8);
  t.u[1] = *(const uint4*)(base + 16 + half * 8);
  return t.v;
}
__device__ __forceinline__ v16bf load_b_frag_lds(const __bf16* base, int half) {
  union { uint4 u[2]; v16bf v; } t;
  t.u[0] = *(const uint4*)(base + half * 8);
  t.u[1] = *(const uint4*)(base + 16 + half * 8);
  return t.v;
}

// ---------------------------------------------------------------------------
// f32 -> bf16 bulk conversion (weights, one pass)
// ---------------------------------------------------------------------------
__global__ __launch_bounds__(256) void k_cvt(const float* __restrict__ s,
                                             __bf16* __restrict__ d, long n)
{
  long i = (long)blockIdx.x * 256 + threadIdx.x;
  if (i < n) d[i] = f2bf(s[i]);
}

// ---------------------------------------------------------------------------
// GEMM: Out[M,Nout] = act(A[M,K] @ W[Nout,K]^T + bias) (+ Res)
// A, W bf16. Block 128 (4 waves), tile 64x128; wave computes 32x64 (8 WMMA
// per K-step). Double-buffered async Global->LDS staging: stage k+1 issued
// before waiting on stage k (ASYNCcnt completes in order -> wait <= 6).
// M multiple of 64; K multiple of 32.
// ---------------------------------------------------------------------------
template<int ACT, int HAS_RES, int OUT_BF>
__global__ __launch_bounds__(128) void k_gemm(
    const __bf16* __restrict__ A, long lda,
    const __bf16* __restrict__ W,
    const float* __restrict__ bias,
    const float* __restrict__ Res, long ldr,
    float* __restrict__ OutF, __bf16* __restrict__ OutH, long ldo,
    int M, int Nout, int K)
{
  __shared__ alignas(16) __bf16 As[2][64][40];   // 80B row stride (16B aligned)
  __shared__ alignas(16) __bf16 Bs[2][128][40];
  const int tid  = threadIdx.x;
  const int lane = tid & 31, wave = tid >> 5;
  const int lr   = lane & 15, half = lane >> 4;
  const int wm   = wave >> 1, wn = wave & 1;      // wave -> 32x64 subtile
  const long m0  = (long)blockIdx.x * 64;
  const long n0  = (long)blockIdx.y * 128;

  v8f acc[2][4] = {};

  auto stage = [&](int k0, int buf) {
#pragma unroll
    for (int t = 0; t < 2; t++) {                 // A: 64x32 bf16 = 256 chunks
      int ci = tid + t * 128;
      int row = ci >> 2, col = (ci & 3) * 8;
      long ar = m0 + row; if (ar > (long)M - 1) ar = M - 1;
      async_copy16(A + ar * lda + k0 + col, &As[buf][row][col]);
    }
#pragma unroll
    for (int t = 0; t < 4; t++) {                 // B: 128x32 bf16 = 512 chunks
      int ci = tid + t * 128;
      int row = ci >> 2, col = (ci & 3) * 8;
      long wr = n0 + row; if (wr > (long)Nout - 1) wr = Nout - 1;
      async_copy16(W + wr * (long)K + k0 + col, &Bs[buf][row][col]);
    }
  };

  const int nk = K >> 5;
  stage(0, 0);
  for (int kb = 0; kb < nk; kb++) {
    const int cur = kb & 1;
    if (kb + 1 < nk) { stage((kb + 1) << 5, cur ^ 1); async_wait<6>(); }
    else             { async_wait<0>(); }
    __syncthreads();

    v16bf a0 = load_a_frag_lds(&As[cur][wm * 32 + lr][0], half);
    v16bf a1 = load_a_frag_lds(&As[cur][wm * 32 + 16 + lr][0], half);
#pragma unroll
    for (int j = 0; j < 4; j++) {
      v16bf bj = load_b_frag_lds(&Bs[cur][wn * 64 + j * 16 + lr][0], half);
      acc[0][j] = WMMA_BF16(a0, bj, acc[0][j]);
      acc[1][j] = WMMA_BF16(a1, bj, acc[1][j]);
    }
    __syncthreads();
  }

#pragma unroll
  for (int i = 0; i < 2; i++)
#pragma unroll
    for (int j = 0; j < 4; j++) {
      long col = n0 + wn * 64 + j * 16 + lr;
      if (col < Nout) {
        float bv = bias[col];
#pragma unroll
        for (int e = 0; e < 8; e++) {
          long rowg = m0 + wm * 32 + i * 16 + e + 8 * half;
          float v = acc[i][j][e] + bv;
          if (ACT) v = 0.5f * v * (1.f + erff(v * 0.70710678118654752f));  // exact GELU
          if (HAS_RES) v += Res[rowg * ldr + col];
          if (OUT_BF) OutH[rowg * ldo + col] = f2bf(v);
          else        OutF[rowg * ldo + col] = v;
        }
      }
    }
}

// ---------------------------------------------------------------------------
// im2col (writes bf16): PA[(b*196+p), k], k = ch*256 + py*16 + px
// ---------------------------------------------------------------------------
__global__ __launch_bounds__(256) void k_im2col(const float* __restrict__ inp,
                                                __bf16* __restrict__ PA)
{
  int k = blockIdx.x * 256 + threadIdx.x;   // < 768
  int p = blockIdx.y, b = blockIdx.z;
  int ch = k >> 8, py = (k >> 4) & 15, px = k & 15;
  int gy = p / 14, gx = p % 14;
  PA[((long)b * NPATCH + p) * 768 + k] =
      f2bf(inp[(((long)b * 3 + ch) * 224 + gy * 16 + py) * 224 + gx * 16 + px]);
}

// assemble x: cls + pos, patches + pos, zero padding rows (f32 residual stream)
__global__ __launch_bounds__(256) void k_assemble(const float* __restrict__ PX,
                                                  const float* __restrict__ cls,
                                                  const float* __restrict__ pos,
                                                  float* __restrict__ X)
{
  int c = blockIdx.x * 256 + threadIdx.x;   // < 768
  int t = blockIdx.y, b = blockIdx.z;
  float v;
  if (t == 0)            v = cls[c] + pos[c];
  else if (t <= NPATCH)  v = PX[((long)b * NPATCH + t - 1) * Cdim + c] + pos[(long)t * Cdim + c];
  else                   v = 0.f;
  X[((long)b * NP + t) * Cdim + c] = v;
}

// ---------------------------------------------------------------------------
// LayerNorm per row (C = 768), f32 in, bf16 out (feeds WMMA GEMMs only)
// ---------------------------------------------------------------------------
__global__ __launch_bounds__(256) void k_layernorm(const float* __restrict__ X,
                                                   const float* __restrict__ g,
                                                   const float* __restrict__ bta,
                                                   __bf16* __restrict__ Y, int Cr)
{
  __shared__ float red[256];
  size_t row = blockIdx.x;
  const float* x = X + row * Cr;
  __bf16* y = Y + row * Cr;
  int tid = threadIdx.x;
  float s = 0.f;
  for (int c = tid; c < Cr; c += 256) s += x[c];
  red[tid] = s; __syncthreads();
  for (int o = 128; o > 0; o >>= 1) { if (tid < o) red[tid] += red[tid + o]; __syncthreads(); }
  float mu = red[0] / Cr; __syncthreads();
  float vs = 0.f;
  for (int c = tid; c < Cr; c += 256) { float d = x[c] - mu; vs += d * d; }
  red[tid] = vs; __syncthreads();
  for (int o = 128; o > 0; o >>= 1) { if (tid < o) red[tid] += red[tid + o]; __syncthreads(); }
  float inv = rsqrtf(red[0] / Cr + 1e-6f);
  for (int c = tid; c < Cr; c += 256) y[c] = f2bf((x[c] - mu) * inv * g[c] + bta[c]);
}

// ---------------------------------------------------------------------------
// Attention scores: ATT[b,h,n,m] = SCALE * q . k  (q,k bf16; acc f32)
// block 64 (2 waves); each wave one 16x16 (n,m) tile; K = 64 (2 WMMA steps)
// ---------------------------------------------------------------------------
__global__ __launch_bounds__(64) void k_scores(const __bf16* __restrict__ QKV,
                                               float* __restrict__ ATT)
{
  int lane = threadIdx.x & 31, wave = threadIdx.x >> 5;
  int lr = lane & 15, half = lane >> 4;
  int bh = blockIdx.z, b = bh / NH, h = bh % NH;
  int n0 = blockIdx.y * 16;
  int m0 = blockIdx.x * 32 + wave * 16;
  const __bf16* qrow = QKV + ((long)b * NP + n0 + lr) * C3 + h * DH;
  const __bf16* krow = QKV + ((long)b * NP + m0 + lr) * C3 + Cdim + h * DH;
  v8f acc = {};
#pragma unroll
  for (int k0 = 0; k0 < DH; k0 += 32) {
    v16bf a  = load_a_frag_bf(qrow, k0, half);
    v16bf bb = load_b_frag_bf(krow, k0, half);
    acc = WMMA_BF16(a, bb, acc);
  }
  float* op = ATT + ((long)bh * NP + n0) * NP + m0;
#pragma unroll
  for (int e = 0; e < 8; e++) op[(long)(e + 8 * half) * NP + lr] = acc[e] * 0.125f;
}

// V transpose cache: VT[b,h,d,m] = QKV[b,m, 2C + h*64 + d]
// One thread per (b,h,m): reads one 128B line, writes coalesced across m.
__global__ __launch_bounds__(256) void k_vtrans(const __bf16* __restrict__ QKV,
                                                __bf16* __restrict__ VT)
{
  int m = threadIdx.x;                    // 0..255 (guard to NP)
  int h = blockIdx.y, b = blockIdx.z;
  if (m >= NP) return;
  const __bf16* src = QKV + ((long)b * NP + m) * C3 + 2 * Cdim + h * DH;
  union { uint4 u[8]; __bf16 e[64]; } t;
#pragma unroll
  for (int i = 0; i < 8; i++) t.u[i] = *(const uint4*)(src + i * 8);
  __bf16* dst = VT + ((long)(b * NH + h) * DH) * (long)NP + m;
#pragma unroll
  for (int d = 0; d < 64; d++) dst[(long)d * NP] = t.e[d];
}

// pre-softmax ReLU row sums (for uncertainty), one block per (b,h,n) row
__global__ __launch_bounds__(256) void k_relurow(const float* __restrict__ ATT,
                                                 float* __restrict__ RS, int Nreal)
{
  __shared__ float red[256];
  size_t row = blockIdx.x;
  const float* a = ATT + row * NP;
  int tid = threadIdx.x;
  float s = 0.f;
  for (int m = tid; m < Nreal; m += 256) s += fmaxf(a[m], 0.f);
  red[tid] = s; __syncthreads();
  for (int o = 128; o > 0; o >>= 1) { if (tid < o) red[tid] += red[tid + o]; __syncthreads(); }
  if (tid == 0) RS[row] = red[0];
}

// unc[b,p] = sum_h N/(N + RS[b,h,1+p])
__global__ __launch_bounds__(256) void k_unc(const float* __restrict__ RS,
                                             float* __restrict__ UNC)
{
  int i = blockIdx.x * 256 + threadIdx.x;
  if (i >= BATCH * NPATCH) return;
  int b = i / NPATCH, p = i % NPATCH, n = 1 + p;
  float u = 0.f;
  for (int h = 0; h < NH; h++)
    u += (float)N0 / ((float)N0 + RS[(size_t)(b * NH + h) * NP + n]);
  UNC[i] = u;
}

// softmax: reads f32 scores, writes normalized bf16 probs (pads zeroed)
__global__ __launch_bounds__(256) void k_softmax(float* __restrict__ ATT,
                                                 __bf16* __restrict__ ATT16, int Nreal)
{
  __shared__ float red[256];
  size_t row = blockIdx.x;
  int n = (int)(row % NP);
  float* a = ATT + row * NP;
  __bf16* p = ATT16 + row * NP;
  int tid = threadIdx.x;
  __bf16 z; { unsigned short zs = 0; __builtin_memcpy(&z, &zs, 2); }
  if (n >= Nreal) { for (int m = tid; m < NP; m += 256) p[m] = z; return; }
  float mx = -3.0e38f;
  for (int m = tid; m < Nreal; m += 256) mx = fmaxf(mx, a[m]);
  red[tid] = mx; __syncthreads();
  for (int o = 128; o > 0; o >>= 1) { if (tid < o) red[tid] = fmaxf(red[tid], red[tid + o]); __syncthreads(); }
  mx = red[0]; __syncthreads();
  float s = 0.f;
  for (int m = tid; m < Nreal; m += 256) { float e = expf(a[m] - mx); a[m] = e; s += e; }
  red[tid] = s; __syncthreads();
  for (int o = 128; o > 0; o >>= 1) { if (tid < o) red[tid] += red[tid + o]; __syncthreads(); }
  float inv = 1.0f / red[0];
  for (int m = tid; m < Nreal; m += 256) p[m] = f2bf(a[m] * inv);
  for (int m = Nreal + tid; m < NP; m += 256) p[m] = z;
}

// O[b,n,h*64+d] = sum_m P[b,h,n,m] * VT[b,h,d,m]; all-bf16 vector loads
__global__ __launch_bounds__(128) void k_attnv(const __bf16* __restrict__ ATT16,
                                               const __bf16* __restrict__ VT,
                                               __bf16* __restrict__ O)
{
  int lane = threadIdx.x & 31, wave = threadIdx.x >> 5;
  int lr = lane & 15, half = lane >> 4;
  int b = blockIdx.z, h = blockIdx.y;
  int n0 = blockIdx.x * 16;
  int d0 = wave * 16;
  const __bf16* arow = ATT16 + (((long)b * NH + h) * NP + n0 + lr) * NP;
  const __bf16* vrow = VT + ((long)(b * NH + h) * DH + d0 + lr) * (long)NP;
  v8f acc = {};
#pragma unroll
  for (int k0 = 0; k0 < NP; k0 += 32) {
    v16bf a  = load_a_frag_bf(arow, k0, half);
    v16bf bb = load_b_frag_bf(vrow, k0, half);
    acc = WMMA_BF16(a, bb, acc);
  }
  __bf16* op = O + ((long)b * NP + n0) * Cdim + h * DH + d0;
#pragma unroll
  for (int e = 0; e < 8; e++) op[(long)(e + 8 * half) * Cdim + lr] = f2bf(acc[e]);
}

// ---------------------------------------------------------------------------
// Token selection: rank tokens by uncertainty (keep KSEL smallest, stable ties)
// ---------------------------------------------------------------------------
__global__ __launch_bounds__(256) void k_rank(const float* __restrict__ UNC,
                                              int* __restrict__ KIDX,
                                              int* __restrict__ NIDX)
{
  __shared__ float u[NPATCH];
  int b = blockIdx.x, tid = threadIdx.x;
  if (tid < NPATCH) u[tid] = UNC[b * NPATCH + tid];
  __syncthreads();
  if (tid < NPATCH) {
    float ui = u[tid];
    int rank = 0;
    for (int j = 0; j < NPATCH; j++) {
      float uj = u[j];
      rank += (uj < ui) || (uj == ui && j < tid);
    }
    if (rank < KSEL) KIDX[b * KSEL + rank] = tid;
    else             NIDX[b * NREJ + (rank - KSEL)] = tid;
  }
}

// build pruned token sequence: [cls, kept..., rst_mean, zeros...]
__global__ __launch_bounds__(256) void k_gather(const float* __restrict__ X,
                                                const int* __restrict__ KIDX,
                                                const int* __restrict__ NIDX,
                                                float* __restrict__ XN)
{
  int c = blockIdx.x * 256 + threadIdx.x;   // < 768
  int t = blockIdx.y, b = blockIdx.z;
  float v;
  if (t == 0) {
    v = X[((long)b * NP) * Cdim + c];
  } else if (t <= KSEL) {
    int p = KIDX[b * KSEL + t - 1];
    v = X[((long)b * NP + 1 + p) * Cdim + c];
  } else if (t == KSEL + 1) {
    float s = 0.f;
    for (int j = 0; j < NREJ; j++)
      s += X[((long)b * NP + 1 + NIDX[b * NREJ + j]) * Cdim + c];
    v = s / (float)NREJ;
  } else {
    v = 0.f;
  }
  XN[((long)b * NP + t) * Cdim + c] = v;
}

// ---------------------------------------------------------------------------
extern "C" void kernel_launch(void* const* d_in, const int* in_sizes, int n_in,
                              void* d_out, int out_size, void* d_ws, size_t ws_size,
                              hipStream_t stream)
{
  const float* inputs  = (const float*)d_in[0];
  const float* patch_w = (const float*)d_in[1];
  const float* patch_b = (const float*)d_in[2];
  const float* cls_tok = (const float*)d_in[3];
  const float* pos_emb = (const float*)d_in[4];
  const float* ln1_g   = (const float*)d_in[5];
  const float* ln1_b   = (const float*)d_in[6];
  const float* qkv_w   = (const float*)d_in[7];
  const float* qkv_b   = (const float*)d_in[8];
  const float* proj_w  = (const float*)d_in[9];
  const float* proj_b  = (const float*)d_in[10];
  const float* ln2_g   = (const float*)d_in[11];
  const float* ln2_b   = (const float*)d_in[12];
  const float* fc1_w   = (const float*)d_in[13];
  const float* fc1_b   = (const float*)d_in[14];
  const float* fc2_w   = (const float*)d_in[15];
  const float* fc2_b   = (const float*)d_in[16];
  const float* norm_g  = (const float*)d_in[17];
  const float* norm_b  = (const float*)d_in[18];
  const float* head_w  = (const float*)d_in[19];
  const float* head_b  = (const float*)d_in[20];
  float* outp = (float*)d_out;

  // ---- workspace carve (256B aligned) ----
  char* wsb = (char*)d_ws;
  size_t off = 0;
  auto carve = [&](size_t bytes) -> void* {
    void* p = wsb + off; off += (bytes + 255) & ~(size_t)255; return p;
  };
  float*  XA    = (float*) carve(4ull * BATCH * NP * Cdim);
  float*  XB    = (float*) carve(4ull * BATCH * NP * Cdim);
  float*  ATT   = (float*) carve(4ull * BATCH * NH * NP * NP);
  __bf16* ATT16 = (__bf16*)carve(2ull * BATCH * NH * NP * NP);
  __bf16* VT16  = (__bf16*)carve(2ull * BATCH * NH * DH * NP);
  __bf16* HB16  = (__bf16*)carve(2ull * BATCH * NP * Cdim);
  __bf16* OB16  = (__bf16*)carve(2ull * BATCH * NP * Cdim);
  __bf16* QKV16 = (__bf16*)carve(2ull * BATCH * NP * C3);
  __bf16* H2_16 = (__bf16*)carve(2ull * BATCH * NP * C4);
  float*  RS    = (float*) carve(4ull * BATCH * NH * NP);
  float*  UNC   = (float*) carve(4ull * BATCH * NPATCH);
  int*    KIDX  = (int*)   carve(4ull * BATCH * KSEL);
  int*    NIDX  = (int*)   carve(4ull * BATCH * NREJ);
  __bf16* WQ16  = (__bf16*)carve(2ull * 12 * C3 * Cdim);
  __bf16* WPJ16 = (__bf16*)carve(2ull * 12 * Cdim * Cdim);
  __bf16* WF116 = (__bf16*)carve(2ull * 12 * C4 * Cdim);
  __bf16* WF216 = (__bf16*)carve(2ull * 12 * Cdim * C4);
  __bf16* WPT16 = (__bf16*)carve(2ull * Cdim * 768);
  __bf16* WHD16 = (__bf16*)carve(2ull * 100 * Cdim);
  __bf16* PA16  = H2_16;  // alias: patch phase precedes layer loop
  float*  PX    = ATT;    // alias: ditto

  // ---- one-time (per call) weight conversion to bf16 ----
  auto cvt = [&](const float* s, __bf16* d, size_t n) {
    k_cvt<<<(unsigned)((n + 255) / 256), 256, 0, stream>>>(s, d, (long)n);
  };
  cvt(qkv_w,  WQ16,  (size_t)12 * C3 * Cdim);
  cvt(proj_w, WPJ16, (size_t)12 * Cdim * Cdim);
  cvt(fc1_w,  WF116, (size_t)12 * C4 * Cdim);
  cvt(fc2_w,  WF216, (size_t)12 * Cdim * C4);
  cvt(patch_w, WPT16, (size_t)Cdim * 768);
  cvt(head_w,  WHD16, (size_t)100 * Cdim);

  // GEMM dispatch helpers (tile 64 x 128)
  auto gemm_f32 = [&](const __bf16* A, long lda, const __bf16* W, const float* bias,
                      float* Out, long ldo, int M, int Nout, int K) {
    k_gemm<0, 0, 0><<<dim3(M / 64, (Nout + 127) / 128), 128, 0, stream>>>(
        A, lda, W, bias, nullptr, 0, Out, nullptr, ldo, M, Nout, K);
  };
  auto gemm_res = [&](const __bf16* A, long lda, const __bf16* W, const float* bias,
                      const float* Res, float* Out, long ldo, int M, int Nout, int K) {
    k_gemm<0, 1, 0><<<dim3(M / 64, (Nout + 127) / 128), 128, 0, stream>>>(
        A, lda, W, bias, Res, ldo, Out, nullptr, ldo, M, Nout, K);
  };
  auto gemm_bf = [&](const __bf16* A, long lda, const __bf16* W, const float* bias,
                     __bf16* Out, long ldo, int M, int Nout, int K, bool gelu) {
    if (gelu)
      k_gemm<1, 0, 1><<<dim3(M / 64, (Nout + 127) / 128), 128, 0, stream>>>(
          A, lda, W, bias, nullptr, 0, nullptr, Out, ldo, M, Nout, K);
    else
      k_gemm<0, 0, 1><<<dim3(M / 64, (Nout + 127) / 128), 128, 0, stream>>>(
          A, lda, W, bias, nullptr, 0, nullptr, Out, ldo, M, Nout, K);
  };

  // ---- patch embedding ----
  k_im2col<<<dim3(3, NPATCH, BATCH), 256, 0, stream>>>(inputs, PA16);
  gemm_f32(PA16, 768, WPT16, patch_b, PX, Cdim, BATCH * NPATCH, Cdim, 768);
  k_assemble<<<dim3(3, NP, BATCH), 256, 0, stream>>>(PX, cls_tok, pos_emb, XA);

  float* X = XA; float* XALT = XB;
  for (int n = 0; n < 12; n++) {
    int Nreal = (n <= 3) ? N0 : N1;  // pruning happens after layer-3 attention

    k_layernorm<<<Mrows, 256, 0, stream>>>(X, ln1_g + n * Cdim, ln1_b + n * Cdim, HB16, Cdim);
    gemm_bf(HB16, Cdim, WQ16 + (size_t)n * C3 * Cdim, qkv_b + n * C3,
            QKV16, C3, Mrows, C3, Cdim, false);

    k_scores<<<dim3(NP / 32, NP / 16, BATCH * NH), 64, 0, stream>>>(QKV16, ATT);
    k_vtrans<<<dim3(1, NH, BATCH), 256, 0, stream>>>(QKV16, VT16);

    if (n == 3) {  // uncertainty from pre-softmax scores
      k_relurow<<<BATCH * NH * NP, 256, 0, stream>>>(ATT, RS, N0);
      k_unc<<<(BATCH * NPATCH + 255) / 256, 256, 0, stream>>>(RS, UNC);
    }

    k_softmax<<<BATCH * NH * NP, 256, 0, stream>>>(ATT, ATT16, Nreal);
    k_attnv<<<dim3(NP / 16, NH, BATCH), 128, 0, stream>>>(ATT16, VT16, OB16);

    gemm_res(OB16, Cdim, WPJ16 + (size_t)n * Cdim * Cdim, proj_b + n * Cdim,
             X, X, Cdim, Mrows, Cdim, Cdim);

    if (n == 3) {
      k_rank<<<BATCH, 256, 0, stream>>>(UNC, KIDX, NIDX);
      k_gather<<<dim3(3, NP, BATCH), 256, 0, stream>>>(X, KIDX, NIDX, XALT);
      float* t = X; X = XALT; XALT = t;
    }

    k_layernorm<<<Mrows, 256, 0, stream>>>(X, ln2_g + n * Cdim, ln2_b + n * Cdim, HB16, Cdim);
    gemm_bf(HB16, Cdim, WF116 + (size_t)n * C4 * Cdim, fc1_b + n * C4,
            H2_16, C4, Mrows, C4, Cdim, true /*gelu*/);
    gemm_res(H2_16, C4, WF216 + (size_t)n * Cdim * C4, fc2_b + n * Cdim,
             X, X, Cdim, Mrows, Cdim, C4);
  }

  // ---- final norm + head (only CLS rows: lda = NP*C strides per batch) ----
  k_layernorm<<<Mrows, 256, 0, stream>>>(X, norm_g, norm_b, HB16, Cdim);
  gemm_f32(HB16, (long)NP * Cdim, WHD16, head_b, outp, 100, BATCH, 100, Cdim);

  (void)in_sizes; (void)n_in; (void)out_size; (void)ws_size;
}
